// GCN_18992345383338
// MI455X (gfx1250) — matrix operand order repaired and checked by reading, compiled, not run on hardware
//
#include <hip/hip_runtime.h>
#include <math.h>

// ---------- problem constants (match reference) ----------
#define NNODE 50000
#define NEDGE 800000
#define F_IN  512
#define HID   128
#define NOUT  40
#define NOUTP 48          // padded to 3 WMMA N-tiles
#define MT    (NNODE/16)  // 3125 M-tiles

typedef __attribute__((ext_vector_type(16))) __bf16 v16bf;
typedef __attribute__((ext_vector_type(8)))  float  v8f;
typedef __attribute__((ext_vector_type(4)))  float  v4f;

// f32 -> bf16, round-to-nearest-even, no backend cvt dependency
static __device__ __forceinline__ __bf16 f2bf(float f) {
    unsigned u = __builtin_bit_cast(unsigned, f);
    unsigned r = u + 0x7FFFu + ((u >> 16) & 1u);
    unsigned short h = (unsigned short)(r >> 16);
    return __builtin_bit_cast(__bf16, h);
}

// ---------- utility fill ----------
__global__ void fill_k(float* __restrict__ p, float v, size_t n) {
    size_t i = (size_t)blockIdx.x * blockDim.x + threadIdx.x;
    if (i < n) p[i] = v;
}

// ---------- degree / norm ----------
__global__ void edge_deg_k(const long long* __restrict__ ei,
                           const float* __restrict__ ew,
                           float* __restrict__ deg) {
    int e = blockIdx.x * blockDim.x + threadIdx.x;
    if (e >= NEDGE) return;
    int col = (int)ei[NEDGE + e];
    unsafeAtomicAdd(&deg[col], ew[e]);
}

__global__ void dinv_k(float* __restrict__ deg) {
    int i = blockIdx.x * blockDim.x + threadIdx.x;
    if (i >= NNODE) return;
    float d = deg[i];
    deg[i] = d > 0.0f ? rsqrtf(d) : 0.0f;
}

// ---------- weight packing into per-lane B-fragment order ----------
// fragment (kt,nt): 32 lanes x 16 bf16.  lane L: n=L%16, kh=L/16; elem e -> K=32*kt+8*kh+e+(e&8)
__global__ void pack_w1_k(const float* __restrict__ W1, __bf16* __restrict__ p) {
    int t = blockIdx.x * blockDim.x + threadIdx.x;       // 16*8*512 = 65536
    if (t >= 16 * 8 * 512) return;
    int frag = t >> 9, r = t & 511;
    int lane = r >> 4, e = r & 15;
    int kt = frag >> 3, nt = frag & 7;
    int kh = lane >> 4, n = lane & 15;
    int K = kt * 32 + kh * 8 + e + (e & 8);
    int Nc = nt * 16 + n;
    p[t] = f2bf(W1[K * HID + Nc]);
}

__global__ void pack_w2_k(const float* __restrict__ W2, __bf16* __restrict__ p) {
    int t = blockIdx.x * blockDim.x + threadIdx.x;       // 4*3*512 = 6144
    if (t >= 4 * 3 * 512) return;
    int frag = t >> 9, r = t & 511;
    int lane = r >> 4, e = r & 15;
    int kt = frag / 3, nt = frag % 3;
    int kh = lane >> 4, n = lane & 15;
    int K = kt * 32 + kh * 8 + e + (e & 8);
    int Nc = nt * 16 + n;
    p[t] = f2bf(Nc < NOUT ? W2[K * NOUT + Nc] : 0.0f);
}

// ---------- GEMM1: h[50000,128] = x[50000,512] @ W1 (bf16 WMMA, f32 acc) ----------
__global__ void __launch_bounds__(256) gemm1_k(const float* __restrict__ x,
                                               const __bf16* __restrict__ w1p,
                                               float* __restrict__ h) {
    int wave = blockIdx.x * 8 + (threadIdx.x >> 5);
    if (wave >= MT) return;                       // wave-uniform guard
    int lane = threadIdx.x & 31;
    int m16 = lane & 15, kh = lane >> 4;
    const float* xrow = x + (size_t)(wave * 16 + m16) * F_IN;

    v8f acc[8];
    #pragma unroll
    for (int i = 0; i < 8; ++i) acc[i] = {};

    for (int kt = 0; kt < 16; ++kt) {
        // ---- issue ALL loads for this K-step up front (one clause, one wait) ----
        const float* ap = xrow + kt * 32 + kh * 8;
        v4f r0 = *(const v4f*)(ap + 0);
        v4f r1 = *(const v4f*)(ap + 4);
        v4f r2 = *(const v4f*)(ap + 16);
        v4f r3 = *(const v4f*)(ap + 20);

        const __bf16* bp = w1p + ((size_t)(kt * 8) * 32 + lane) * 16;
        v16bf bfrag[8];
        #pragma unroll
        for (int nt = 0; nt < 8; ++nt)
            bfrag[nt] = *(const v16bf*)(bp + (size_t)nt * 32 * 16);

        // ---- convert A while loads are in flight ----
        v16bf a;
        #pragma unroll
        for (int j = 0; j < 4; ++j) {
            a[j]      = f2bf(r0[j]);
            a[4 + j]  = f2bf(r1[j]);
            a[8 + j]  = f2bf(r2[j]);
            a[12 + j] = f2bf(r3[j]);
        }

        // ---- back-to-back WMMA chain ----
        #pragma unroll
        for (int nt = 0; nt < 8; ++nt)
            acc[nt] = __builtin_amdgcn_wmma_f32_16x16x32_bf16(
                false, a, false, bfrag[nt], (short)0, acc[nt], false, false);
    }
    float* hb = h + (size_t)(wave * 16 + 8 * kh) * HID + m16;
    #pragma unroll
    for (int nt = 0; nt < 8; ++nt)
        #pragma unroll
        for (int r = 0; r < 8; ++r)
            hb[(size_t)r * HID + nt * 16] = acc[nt][r];
}

// ---------- edge scatter for layer 1: out1[col] += norm * h[row] ----------
__global__ void scatter1_k(const long long* __restrict__ ei,
                           const float* __restrict__ ew,
                           const float* __restrict__ dinv,
                           const float* __restrict__ h,
                           float* __restrict__ out1) {
    int e = blockIdx.x * 8 + (threadIdx.x >> 5);
    if (e >= NEDGE) return;
    int lane = threadIdx.x & 31;
    int row = (int)ei[e];
    int col = (int)ei[NEDGE + e];
    float norm = dinv[row] * ew[e] * dinv[col];
    v4f hv = *(const v4f*)(h + (size_t)row * HID + lane * 4);
    float* ob = out1 + (size_t)col * HID + lane * 4;
    #pragma unroll
    for (int j = 0; j < 4; ++j) unsafeAtomicAdd(ob + j, norm * hv[j]);
}

// ---------- finalize 1: out1 = relu(out1 + dinv^2*h + b1), in place ----------
__global__ void finalize1_k(const float* __restrict__ h,
                            const float* __restrict__ dinv,
                            const float* __restrict__ b1,
                            float* __restrict__ out1) {
    size_t i = (size_t)blockIdx.x * blockDim.x + threadIdx.x;
    if (i >= (size_t)NNODE * HID) return;
    int node = (int)(i >> 7);
    int f = (int)(i & 127);
    float dv = dinv[node];
    float v = out1[i] + dv * dv * h[i] + b1[f];
    out1[i] = v > 0.0f ? v : 0.0f;
}

// ---------- GEMM2: z[50000,48] = out1[50000,128] @ W2pad ----------
__global__ void __launch_bounds__(256) gemm2_k(const float* __restrict__ A,
                                               const __bf16* __restrict__ w2p,
                                               float* __restrict__ z) {
    int wave = blockIdx.x * 8 + (threadIdx.x >> 5);
    if (wave >= MT) return;
    int lane = threadIdx.x & 31;
    int m16 = lane & 15, kh = lane >> 4;
    const float* arow = A + (size_t)(wave * 16 + m16) * HID;

    v8f acc[3];
    acc[0] = {}; acc[1] = {}; acc[2] = {};

    #pragma unroll
    for (int kt = 0; kt < 4; ++kt) {
        const float* ap = arow + kt * 32 + kh * 8;
        v4f r0 = *(const v4f*)(ap + 0);
        v4f r1 = *(const v4f*)(ap + 4);
        v4f r2 = *(const v4f*)(ap + 16);
        v4f r3 = *(const v4f*)(ap + 20);

        const __bf16* bp = w2p + ((size_t)(kt * 3) * 32 + lane) * 16;
        v16bf bfrag[3];
        #pragma unroll
        for (int nt = 0; nt < 3; ++nt)
            bfrag[nt] = *(const v16bf*)(bp + (size_t)nt * 32 * 16);

        v16bf a;
        #pragma unroll
        for (int j = 0; j < 4; ++j) {
            a[j]      = f2bf(r0[j]);
            a[4 + j]  = f2bf(r1[j]);
            a[8 + j]  = f2bf(r2[j]);
            a[12 + j] = f2bf(r3[j]);
        }

        #pragma unroll
        for (int nt = 0; nt < 3; ++nt)
            acc[nt] = __builtin_amdgcn_wmma_f32_16x16x32_bf16(
                false, a, false, bfrag[nt], (short)0, acc[nt], false, false);
    }
    float* zb = z + (size_t)(wave * 16 + 8 * kh) * NOUTP + m16;
    #pragma unroll
    for (int nt = 0; nt < 3; ++nt)
        #pragma unroll
        for (int r = 0; r < 8; ++r)
            zb[(size_t)r * NOUTP + nt * 16] = acc[nt][r];
}

// ---------- edge scatter for layer 2: dout[col] += norm * z[row] (40 cols) ----------
__global__ void scatter2_k(const long long* __restrict__ ei,
                           const float* __restrict__ ew,
                           const float* __restrict__ dinv,
                           const float* __restrict__ z,
                           float* __restrict__ dout) {
    int e = blockIdx.x * 8 + (threadIdx.x >> 5);
    if (e >= NEDGE) return;
    int lane = threadIdx.x & 31;
    int row = (int)ei[e];
    int col = (int)ei[NEDGE + e];
    float norm = dinv[row] * ew[e] * dinv[col];
    const float* zr = z + (size_t)row * NOUTP;
    float* ob = dout + (size_t)col * NOUT;
    unsafeAtomicAdd(ob + lane, norm * zr[lane]);
    if (lane < 8) unsafeAtomicAdd(ob + 32 + lane, norm * zr[32 + lane]);
}

// ---------- finalize 2: add self-loop + b2, log_softmax over 40, one wave/node ----------
__global__ void finalize2_k(const float* __restrict__ z,
                            const float* __restrict__ dinv,
                            const float* __restrict__ b2,
                            float* __restrict__ dout) {
    int node = blockIdx.x * 8 + (threadIdx.x >> 5);
    if (node >= NNODE) return;
    int lane = threadIdx.x & 31;
    float dv = dinv[node], d2 = dv * dv;
    const float* zr = z + (size_t)node * NOUTP;
    float* orow = dout + (size_t)node * NOUT;

    float v0 = orow[lane] + d2 * zr[lane] + b2[lane];
    bool has2 = lane < 8;
    float v1 = -INFINITY;
    if (has2) v1 = orow[32 + lane] + d2 * zr[32 + lane] + b2[32 + lane];

    float m = fmaxf(v0, v1);
    #pragma unroll
    for (int o = 16; o > 0; o >>= 1) m = fmaxf(m, __shfl_xor(m, o, 32));
    float s = expf(v0 - m) + (has2 ? expf(v1 - m) : 0.0f);
    #pragma unroll
    for (int o = 16; o > 0; o >>= 1) s += __shfl_xor(s, o, 32);
    float l = m + logf(s);

    orow[lane] = v0 - l;
    if (has2) orow[32 + lane] = v1 - l;
}

// ---------- host launcher ----------
extern "C" void kernel_launch(void* const* d_in, const int* in_sizes, int n_in,
                              void* d_out, int out_size, void* d_ws, size_t ws_size,
                              hipStream_t stream) {
    const float*     x   = (const float*)d_in[0];
    const long long* ei  = (const long long*)d_in[1];   // int64 [2, E]
    const float*     ew  = (const float*)d_in[2];
    const float*     W1  = (const float*)d_in[3];
    const float*     b1  = (const float*)d_in[4];
    const float*     W2  = (const float*)d_in[5];
    const float*     b2  = (const float*)d_in[6];
    float*           out = (float*)d_out;

    // workspace layout (all 256B aligned)
    char* ws = (char*)d_ws;
    float*  h    = (float*)(ws);                                   // 50000*128 f32 = 25,600,000 B
    float*  out1 = (float*)(ws + 25600000);                        // 25,600,000 B
    float*  z    = (float*)(ws + 51200000);                        // 50000*48 f32 = 9,600,000 B
    float*  dinv = (float*)(ws + 60800000);                        // 200,192 B (padded)
    __bf16* w1p  = (__bf16*)(ws + 61000192);                       // 131,072 B
    __bf16* w2p  = (__bf16*)(ws + 61131264);                       // 12,288 B
    (void)in_sizes; (void)n_in; (void)out_size; (void)ws_size;

    // pack weights into WMMA B-fragment order
    pack_w1_k<<<256, 256, 0, stream>>>(W1, w1p);
    pack_w2_k<<<24, 256, 0, stream>>>(W2, w2p);

    // weighted in-degree (self-loop contributes 1.0) -> dinv = rsqrt(deg)
    fill_k<<<(NNODE + 255) / 256, 256, 0, stream>>>(dinv, 1.0f, NNODE);
    edge_deg_k<<<(NEDGE + 255) / 256, 256, 0, stream>>>(ei, ew, dinv);
    dinv_k<<<(NNODE + 255) / 256, 256, 0, stream>>>(dinv);

    // layer 1
    gemm1_k<<<(MT + 7) / 8, 256, 0, stream>>>(x, w1p, h);
    fill_k<<<((size_t)NNODE * HID + 255) / 256, 256, 0, stream>>>(out1, 0.0f, (size_t)NNODE * HID);
    scatter1_k<<<(NEDGE + 7) / 8, 256, 0, stream>>>(ei, ew, dinv, h, out1);
    finalize1_k<<<((size_t)NNODE * HID + 255) / 256, 256, 0, stream>>>(h, dinv, b1, out1);

    // layer 2
    gemm2_k<<<(MT + 7) / 8, 256, 0, stream>>>(out1, w2p, z);
    fill_k<<<((size_t)NNODE * NOUT + 255) / 256, 256, 0, stream>>>(out, 0.0f, (size_t)NNODE * NOUT);
    scatter2_k<<<(NEDGE + 7) / 8, 256, 0, stream>>>(ei, ew, dinv, z, out);
    finalize2_k<<<(NNODE + 7) / 8, 256, 0, stream>>>(z, dinv, b2, out);
}
